// sLSTM_52733608460757
// MI455X (gfx1250) — matrix operand order repaired and checked by reading, not compiled, not run on hardware
//
#include <hip/hip_runtime.h>
#include <hip/hip_bf16.h>

// ---------------- problem constants ----------------
#define BATCH   16
#define TSEQ    2048
#define IDIM    512
#define HDIM    512
#define NLAYER  2
#define KDIM    1024            // IDIM + HDIM (concat x and h in K)
#define NGATE   2048            // 4*HDIM gate columns
#define NTILES  128             // NGATE / 16
#define KTILES  32              // KDIM / 32
#define NBLOCKS 16
#define NTHREADS 256            // 8 wave32 per block -> 128 waves total
#define TOTALTHREADS (NBLOCKS*NTHREADS)

// A-pack per layer: KTILES * 32 lanes * 16 halves = 16384 ushorts = 32 KB
#define APACK_ELEMS (KTILES * 32 * 16)

// ---------------- workspace layout (bytes) ----------------
#define BAR_OFF    0u
#define GATES_OFF  256u                         // f32 [16][2048]  = 131072 B
#define CSTATE_OFF (GATES_OFF + 131072u)        // f32 [2][16][512]=  65536 B
#define APACK_OFF  (CSTATE_OFF + 65536u)        // u16 [2][32][32][16] = 65536 B
#define WPACK_OFF  (APACK_OFF + 65536u)         // u16 [2][128][32][32][16] = 8 MB

typedef __attribute__((ext_vector_type(16))) __bf16         v16bf;
typedef __attribute__((ext_vector_type(16))) unsigned short v16u;
typedef __attribute__((ext_vector_type(8)))  float          v8f;

// float -> bf16, round-to-nearest-even
__device__ __forceinline__ unsigned short f2bf(float f) {
    unsigned int u = __float_as_uint(f);
    u += 0x7FFFu + ((u >> 16) & 1u);
    return (unsigned short)(u >> 16);
}

// Store one (m,k) element of the bf16 A-matrix [16 x KDIM] into the packed
// per-lane WMMA A-fragment layout for v_wmma_f32_16x16x32_bf16:
//   lanes 0-15  hold M=lane,    K in {0..7, 16..23}  (halves 0..15)
//   lanes 16-31 hold M=lane-16, K in {8..15, 24..31}
__device__ __forceinline__ void store_a(unsigned short* apack, int layer,
                                        int m, int k, unsigned short v) {
    int kt   = k >> 5;
    int k_in = k & 31;
    bool g0  = (k_in < 8) | ((k_in >= 16) & (k_in < 24));
    int lane = g0 ? m : (m + 16);
    int h;
    if (g0)  h = (k_in < 8)  ? k_in       : k_in - 8;
    else     h = (k_in < 16) ? k_in - 8   : k_in - 16;
    apack[(((size_t)layer * KTILES + kt) * 32 + lane) * 16 + h] = v;
}

// Device-wide sense-reversing barrier (persistent grid of NBLOCKS blocks).
__device__ __forceinline__ void grid_barrier(unsigned int* bar) {
    __syncthreads();
    if (threadIdx.x == 0) {
        __threadfence();  // make this block's global stores visible
        unsigned int gen = __hip_atomic_load(&bar[1], __ATOMIC_ACQUIRE,
                                             __HIP_MEMORY_SCOPE_AGENT);
        unsigned int arrived = __hip_atomic_fetch_add(&bar[0], 1u,
                                                      __ATOMIC_ACQ_REL,
                                                      __HIP_MEMORY_SCOPE_AGENT);
        if (arrived == NBLOCKS - 1) {
            __hip_atomic_store(&bar[0], 0u, __ATOMIC_RELAXED,
                               __HIP_MEMORY_SCOPE_AGENT);
            __hip_atomic_store(&bar[1], gen + 1u, __ATOMIC_RELEASE,
                               __HIP_MEMORY_SCOPE_AGENT);
        } else {
            while (__hip_atomic_load(&bar[1], __ATOMIC_ACQUIRE,
                                     __HIP_MEMORY_SCOPE_AGENT) == gen) {
                __builtin_amdgcn_s_sleep(2);
            }
        }
    }
    __syncthreads();
}

// ---------------- init: zero barrier, c-state, A-pack ----------------
__global__ void slstm_init(unsigned char* ws) {
    unsigned int i = blockIdx.x * blockDim.x + threadIdx.x;
    if (i < 64u) {
        ((unsigned int*)ws)[i] = 0u;                       // barrier words
    }
    if (i < 32768u) {
        ((unsigned int*)(ws + CSTATE_OFF))[i] = 0u;        // CSTATE (16384) + APACK (16384)
    }
}

// ---------------- prepack weights into WMMA B-fragment layout ----------------
// wpack element i decodes as [l][ntile][ktile][lane][half].
// B-tile (32x16 KxN): lanes 0-15 -> N=lane,    halves = K {0..7, 16..23}
//                     lanes16-31 -> N=lane-16, halves = K {8..15, 24..31}
// B[k][n] = Wcat[n][k], Wcat = [w_ih | w_hh] over K = 0..1023.
__global__ void slstm_prepack(const float* __restrict__ w_ih,
                              const float* __restrict__ w_hh,
                              unsigned char* __restrict__ ws) {
    size_t i = (size_t)blockIdx.x * blockDim.x + threadIdx.x;
    if (i >= (size_t)NLAYER * NTILES * KTILES * 32 * 16) return;
    unsigned short* wpack = (unsigned short*)(ws + WPACK_OFF);
    int h    = (int)(i         & 15);
    int lane = (int)((i >> 4)  & 31);
    int kt   = (int)((i >> 9)  & 31);
    int nt   = (int)((i >> 14) & 127);
    int l    = (int)(i >> 21);
    int ncol = lane & 15;
    int k_in = ((h < 8) ? h : (h + 8)) + ((lane < 16) ? 0 : 8);
    int k    = kt * 32 + k_in;
    int n    = nt * 16 + ncol;
    float w  = (k < IDIM)
                   ? w_ih[((size_t)l * NGATE + n) * IDIM + k]
                   : w_hh[((size_t)l * NGATE + n) * HDIM + (k - IDIM)];
    wpack[i] = f2bf(w);
}

// ---------------- persistent recurrent kernel ----------------
__global__ __launch_bounds__(NTHREADS, 1)
void slstm_persistent(const float* __restrict__ x_in,   // [B,T,I]
                      const float* __restrict__ bias,   // [L,4H]
                      float* __restrict__ out,          // out | h_f | c_f
                      unsigned char* __restrict__ ws) {
    unsigned int*   bar    = (unsigned int*)(ws + BAR_OFF);
    float*          gates  = (float*)(ws + GATES_OFF);
    float*          cstate = (float*)(ws + CSTATE_OFF);
    unsigned short* apack  = (unsigned short*)(ws + APACK_OFF);
    const unsigned short* wpack = (const unsigned short*)(ws + WPACK_OFF);

    // LDS stage for the shared A-pack of the current layer (32 KB of 320 KB)
    __shared__ __align__(128) unsigned short lds_a[APACK_ELEMS];

    const int tid  = threadIdx.x;
    const int gtid = blockIdx.x * NTHREADS + tid;
    const int lane = tid & 31;
    const int wave = blockIdx.x * (NTHREADS / 32) + (tid >> 5);  // N-tile id 0..127

    float* hf_out = out + (size_t)BATCH * TSEQ * HDIM;
    float* cf_out = hf_out + (size_t)NLAYER * BATCH * HDIM;

    const int ncol  = lane & 15;
    const int mbase = (lane < 16) ? 0 : 8;

    for (int t = 0; t < TSEQ; ++t) {
        // ---- phase 1: pack x_t (f32->bf16) into layer-0 A fragments, K 0..511
        #pragma unroll
        for (int r = 0; r < 2; ++r) {
            int idx = gtid + r * TOTALTHREADS;      // 8192 elements (b,j)
            int b = idx >> 9, j = idx & 511;
            float xv = x_in[((size_t)b * TSEQ + t) * IDIM + j];
            store_a(apack, 0, b, j, f2bf(xv));
        }
        grid_barrier(bar);

        for (int l = 0; l < NLAYER; ++l) {
            // ---- phase 2a: async-copy shared A-pack (32 KB) global -> LDS.
            // 2048 B128 transfers, 8 per thread, tracked by ASYNCcnt.
            {
                const unsigned short* asrc = apack + (size_t)l * APACK_ELEMS;
                #pragma unroll
                for (int r = 0; r < 8; ++r) {
                    int c = tid + r * NTHREADS;             // B128 chunk id
                    unsigned int lds_addr =
                        (unsigned int)(size_t)(&lds_a[c * 8]);
                    unsigned long long gaddr =
                        (unsigned long long)(const void*)(asrc + c * 8);
                    asm volatile(
                        "global_load_async_to_lds_b128 %0, %1, off"
                        :
                        : "v"(lds_addr), "v"(gaddr)
                        : "memory");
                }
                asm volatile("s_wait_asynccnt 0x0" ::: "memory");
            }
            __syncthreads();

            // ---- phase 2b: GEMM  gates[16 x 2048] = A[16 x 1024] x B[1024 x 2048]
            // A fragments from LDS (ds_load_b128), B fragments from L2-resident
            // packed weights (global_load_b128 + prefetch).
            v8f acc = {};
            const unsigned short* ap = lds_a + (size_t)lane * 16;
            const unsigned short* wp =
                wpack + ((size_t)(l * NTILES + wave)) * KTILES * 512 +
                (size_t)lane * 16;
            #pragma unroll 8
            for (int kt = 0; kt < KTILES; ++kt) {
                __builtin_prefetch((const void*)(wp + (kt + 4) * 512), 0, 1);
                v16u  au = *(const v16u*)(ap + (size_t)kt * 512);
                v16u  bu = *(const v16u*)(wp + (size_t)kt * 512);
                v16bf a  = __builtin_bit_cast(v16bf, au);
                v16bf b  = __builtin_bit_cast(v16bf, bu);
                acc = __builtin_amdgcn_wmma_f32_16x16x32_bf16(
                          false, a, false, b, (short)0, acc, false, false);
            }
            int   n  = wave * 16 + ncol;
            float bv = bias[l * NGATE + n];
            #pragma unroll
            for (int r = 0; r < 8; ++r) {
                gates[(size_t)(mbase + r) * NGATE + n] = acc[r] + bv;
            }
            grid_barrier(bar);

            // ---- phase 3: gate nonlinearities + state update
            #pragma unroll
            for (int r = 0; r < 2; ++r) {
                int idx = gtid + r * TOTALTHREADS;  // 8192 elements (b,j)
                int b = idx >> 9, j = idx & 511;
                const float* g = gates + (size_t)b * NGATE;
                float ig = g[j];
                float fg = g[HDIM + j];
                float gg = g[2 * HDIM + j];
                float og = g[3 * HDIM + j];
                size_t ci = ((size_t)l * BATCH + b) * HDIM + j;
                float c_new = __expf(fg) * cstate[ci] + __expf(ig) * tanhf(gg);
                float h_new = (1.0f / (1.0f + __expf(-og))) * tanhf(c_new);
                cstate[ci] = c_new;
                unsigned short hb = f2bf(h_new);
                store_a(apack, l, b, HDIM + j, hb);        // h-half for next step
                if (l + 1 < NLAYER) {
                    store_a(apack, l + 1, b, j, hb);       // x-half of next layer
                } else {
                    out[((size_t)b * TSEQ + t) * HDIM + j] = h_new;
                }
                if (t == TSEQ - 1) {
                    hf_out[ci] = h_new;
                    cf_out[ci] = c_new;
                }
            }
            grid_barrier(bar);
        }
    }
}

// ---------------- launch ----------------
extern "C" void kernel_launch(void* const* d_in, const int* in_sizes, int n_in,
                              void* d_out, int out_size, void* d_ws, size_t ws_size,
                              hipStream_t stream) {
    const float* x    = (const float*)d_in[0];   // [B,T,I]
    const float* w_ih = (const float*)d_in[1];   // [L,4H,I]
    const float* w_hh = (const float*)d_in[2];   // [L,4H,H]
    const float* bias = (const float*)d_in[3];   // [L,4H]
    float*       out  = (float*)d_out;
    unsigned char* ws = (unsigned char*)d_ws;

    hipLaunchKernelGGL(slstm_init, dim3(129), dim3(256), 0, stream, ws);

    const int pack_elems = NLAYER * NTILES * KTILES * 32 * 16;  // 4,194,304
    hipLaunchKernelGGL(slstm_prepack, dim3(pack_elems / 256), dim3(256), 0,
                       stream, w_ih, w_hh, ws);

    hipLaunchKernelGGL(slstm_persistent, dim3(NBLOCKS), dim3(NTHREADS), 0,
                       stream, x, bias, out, ws);
}
